// FraudHGNN_36498632081921
// MI455X (gfx1250) — compile-verified, matrix operand order checked by reading, not run on hardware
//
#include <hip/hip_runtime.h>
#include <hip/hip_bf16.h>

typedef __bf16 bf16_t;
typedef __attribute__((ext_vector_type(16))) __bf16 v16bf;
typedef __attribute__((ext_vector_type(8)))  __bf16 v8bf;
typedef __attribute__((ext_vector_type(8)))  float  v8f;

#define NTX   200000
#define NCARD 20000
#define NDEV  5000

__device__ __forceinline__ bf16_t to_bf16(float x) {
    unsigned u = __builtin_bit_cast(unsigned, x);
    unsigned r = (u + 0x7FFFu + ((u >> 16) & 1u)) >> 16;
    unsigned short s = (unsigned short)r;
    return __builtin_bit_cast(bf16_t, s);
}

__device__ __forceinline__ float gelu_f(float x) {
    return 0.5f * x * (1.0f + erff(x * 0.70710678118654752440f));
}

// monotone encoding of float for unsigned atomicMax
__device__ __forceinline__ unsigned enc_f(float x) {
    unsigned b = __builtin_bit_cast(unsigned, x);
    return b ^ (unsigned)(((int)b >> 31) | 0x80000000);
}
__device__ __forceinline__ float dec_f(unsigned u) {
    unsigned b = (u & 0x80000000u) ? (u ^ 0x80000000u) : ~u;
    return __builtin_bit_cast(float, b);
}
#define ENC_NEGINF 0x007FFFFFu  // enc(-inf)

// ---------------------------------------------------------------- utilities
__global__ void fill_u32(unsigned* __restrict__ p, unsigned v, int n) {
    int i = blockIdx.x * blockDim.x + threadIdx.x;
    if (i < n) p[i] = v;
}

__global__ void cvt_bf16(const float* __restrict__ src, bf16_t* __restrict__ dst, int n) {
    int i = blockIdx.x * blockDim.x + threadIdx.x;
    if (i < n) dst[i] = to_bf16(src[i]);
}

// Pack f32 weight [128, N] into WMMA B-fragment order: [N/16][kb=4][lane=32][e=16] bf16.
__global__ void pack_B(const float* __restrict__ src, bf16_t* __restrict__ dst, int N) {
    int idx = blockIdx.x * blockDim.x + threadIdx.x;
    if (idx >= N * 128) return;
    int nt   = idx >> 11;
    int r    = idx & 2047;
    int kb   = r >> 9;
    int lane = (r >> 4) & 31;
    int e    = r & 15;
    int n  = nt * 16 + (lane & 15);
    int kg = lane >> 4;
    int k  = kb * 32 + (e < 8 ? kg * 8 + e : 16 + kg * 8 + (e - 8));
    dst[idx] = to_bf16(src[(size_t)k * N + n]);
}

// Same packing, source is [nmat, 2, 64, 64] per-head matrices expanded
// block-diagonally to 128x128 (head h occupies rows/cols h*64..h*64+63).
__global__ void pack_blockdiag_B(const float* __restrict__ src, bf16_t* __restrict__ dst, int nmat) {
    int idx = blockIdx.x * blockDim.x + threadIdx.x;
    if (idx >= nmat * 16384) return;
    int mat = idx >> 14;
    int r0  = idx & 16383;
    int nt   = r0 >> 11;
    int r    = r0 & 2047;
    int kb   = r >> 9;
    int lane = (r >> 4) & 31;
    int e    = r & 15;
    int n  = nt * 16 + (lane & 15);
    int kg = lane >> 4;
    int k  = kb * 32 + (e < 8 ? kg * 8 + e : 16 + kg * 8 + (e - 8));
    float v = 0.0f;
    if ((k >> 6) == (n >> 6))
        v = src[(((size_t)mat * 2 + (k >> 6)) * 64 + (k & 63)) * 64 + (n & 63)];
    dst[idx] = to_bf16(v);
}

// h[n,j] = gelu(x[n] * W[j] + b[j]) for 1-wide inputs (card/dev); f32 + bf16 out
__global__ void lin_small(const float* __restrict__ x, const float* __restrict__ W,
                          const float* __restrict__ b, float* __restrict__ out,
                          bf16_t* __restrict__ outb, int N) {
    int idx = blockIdx.x * blockDim.x + threadIdx.x;
    if (idx >= N * 128) return;
    int n = idx >> 7, j = idx & 127;
    float v = gelu_f(x[n] * W[j] + b[j]);
    out[idx]  = v;
    outb[idx] = to_bf16(v);
}

// ---------------------------------------------------------------- WMMA GEMM
// C[M,N] = A_bf16[M,128] @ Bpacked[128,N] (+bias), fused epilogues.
// Block: 64 rows x full N. A tile staged in LDS once; 8 waves = 4 m-tiles x
// 2 wave-columns; A fragments preloaded to registers. B fragments are
// software-pipelined: next n-tile's 4 fragments are prefetched before the
// current tile's 4 chained WMMAs issue.
// EPI 0: C = acc (no bias)                                 (kt / vt)
// EPI 1: v = gelu(acc+bias)*decay  -> C f32, Cb bf16       (tx input proj)
// EPI 2: KQV split: n<128 -> Cb[m*256+n] (K, bf16)
//                   n<256 -> C[m*128+n-128] (Q, f32)
//                   else  -> Cb[m*256+n-128] (V, bf16)
// EPI 3: a = acc+bout; sg = sigmoid(skip); nh = sg*a+(1-sg)*h_old;
//        res = gelu(nh)+h_old -> C f32, Cb bf16            (node update)
template <int EPI>
__global__ __launch_bounds__(256) void wmma_gemm_bf16(
    const bf16_t* __restrict__ A, int lda,
    const bf16_t* __restrict__ Bp,
    const float* __restrict__ bias,
    float* __restrict__ C,
    bf16_t* __restrict__ Cb,
    const float* __restrict__ hold,
    const float* __restrict__ decay,
    const float* __restrict__ skipv,
    int M, int N)
{
    __shared__ __align__(16) bf16_t As[64 * 128];
    const int tid  = threadIdx.x;
    const int wave = tid >> 5, lane = tid & 31;
    const int kg = lane >> 4, lr = lane & 15;
    const int m_blk = blockIdx.x * 64;

    // stage A tile: 64 rows x 128 cols bf16; 4 threads/row x 32 elements
    {
        int r = tid >> 2;
        int c = (tid & 3) * 32;
        int gm = m_blk + r;
        if (gm >= M) gm = M - 1;
        const bf16_t* src = A + (size_t)gm * lda + c;
#pragma unroll
        for (int j = 0; j < 4; ++j)
            *(uint4*)(&As[r * 128 + c + j * 8]) = *(const uint4*)(src + j * 8);
    }
    __syncthreads();

    const int mt   = wave >> 1;          // 0..3
    const int mrow = mt * 16 + lr;       // A-fragment row in LDS
    const int NT   = N >> 4;

    // preload all 4 A fragments into registers (reused across all n-tiles)
    v16bf afr[4];
#pragma unroll
    for (int kb = 0; kb < 4; ++kb) {
        v8bf lo = *(const v8bf*)(&As[mrow * 128 + kb * 32 + kg * 8]);
        v8bf hi = *(const v8bf*)(&As[mrow * 128 + kb * 32 + 16 + kg * 8]);
        afr[kb] = __builtin_shufflevector(lo, hi, 0, 1, 2, 3, 4, 5, 6, 7,
                                          8, 9, 10, 11, 12, 13, 14, 15);
    }

    float sg = 0.0f;
    if (EPI == 3) sg = 1.0f / (1.0f + expf(-skipv[0]));
    const bool fullM = (m_blk + 64 <= M);

    const int nt0 = wave & 1;
    // prologue: fetch first n-tile's B fragments
    v16bf bfr[4];
#pragma unroll
    for (int kb = 0; kb < 4; ++kb)
        bfr[kb] = *(const v16bf*)(Bp + (((size_t)nt0 * 4 + kb) * 32 + lane) * 16);

    for (int nt = nt0; nt < NT; nt += 2) {
        // prefetch next n-tile's B fragments before consuming current ones
        v16bf bnx[4];
        const int ntn = nt + 2;
        if (ntn < NT) {
#pragma unroll
            for (int kb = 0; kb < 4; ++kb)
                bnx[kb] = *(const v16bf*)(Bp + (((size_t)ntn * 4 + kb) * 32 + lane) * 16);
        }

        v8f acc = {0.f, 0.f, 0.f, 0.f, 0.f, 0.f, 0.f, 0.f};
#pragma unroll
        for (int kb = 0; kb < 4; ++kb)
            acc = __builtin_amdgcn_wmma_f32_16x16x32_bf16(false, afr[kb], false, bfr[kb],
                                                          (short)0, acc, false, false);

        const int n  = nt * 16 + lr;
        const int m0 = m_blk + mt * 16;
        const float bv = (EPI == 0) ? 0.0f : bias[n];

        // epilogue; guard=false path is branch-free (all blocks but the last)
        auto epilogue = [&](bool guard) {
            if (EPI == 2) {
                // column-range branch is wave-uniform per n-tile
                if (n < 128) {
#pragma unroll
                    for (int v = 0; v < 8; ++v) {
                        int m = m0 + v + 8 * kg;
                        if (guard && m >= M) continue;
                        Cb[(size_t)m * 256 + n] = to_bf16(acc[v] + bv);
                    }
                } else if (n < 256) {
#pragma unroll
                    for (int v = 0; v < 8; ++v) {
                        int m = m0 + v + 8 * kg;
                        if (guard && m >= M) continue;
                        C[(size_t)m * 128 + (n - 128)] = acc[v] + bv;
                    }
                } else {
#pragma unroll
                    for (int v = 0; v < 8; ++v) {
                        int m = m0 + v + 8 * kg;
                        if (guard && m >= M) continue;
                        Cb[(size_t)m * 256 + (n - 128)] = to_bf16(acc[v] + bv);
                    }
                }
            } else {
#pragma unroll
                for (int v = 0; v < 8; ++v) {
                    int m = m0 + v + 8 * kg;
                    if (guard && m >= M) continue;
                    float val = acc[v] + bv;
                    if (EPI == 0) {
                        C[(size_t)m * N + n] = val;
                    } else if (EPI == 1) {
                        val = gelu_f(val) * decay[m];
                        C[(size_t)m * 128 + n]  = val;
                        Cb[(size_t)m * 128 + n] = to_bf16(val);
                    } else {  // EPI 3
                        float ho = hold[(size_t)m * 128 + n];
                        float nh = sg * val + (1.0f - sg) * ho;
                        float res = gelu_f(nh) + ho;
                        C[(size_t)m * 128 + n]  = res;
                        Cb[(size_t)m * 128 + n] = to_bf16(res);
                    }
                }
            }
        };
        if (fullM) epilogue(false);
        else       epilogue(true);

#pragma unroll
        for (int kb = 0; kb < 4; ++kb) bfr[kb] = bnx[kb];
    }
}

// ------------------------------------------------- segment softmax (card/dev)
__global__ __launch_bounds__(256) void edge_logits_max(
    const float* __restrict__ Qdst,    // [Ndst,128] f32
    const float* __restrict__ kt,      // [NTX,128] f32
    const int* __restrict__ esrc, const int* __restrict__ edst,
    const float* __restrict__ prel, int p_off,
    float* __restrict__ elog, unsigned* __restrict__ menc, int E)
{
    int i = blockIdx.x * 8 + (threadIdx.x >> 5);
    if (i >= E) return;
    int lane = threadIdx.x & 31;
    int head = lane >> 4, t = lane & 15;
    int s = esrc[i], d = edst[i];
    const float* q  = Qdst + (size_t)d * 128 + head * 64;
    const float* kr = kt + (size_t)s * 128 + head * 64;
    float p = 0.0f;
#pragma unroll
    for (int jj = 0; jj < 64; jj += 16) p += q[jj + t] * kr[jj + t];
#pragma unroll
    for (int mk = 1; mk < 16; mk <<= 1) p += __shfl_xor(p, mk, 32);
    float logit = p * prel[p_off + head] * 0.125f;  // 1/sqrt(64)
    if (t == 0) {
        elog[i * 2 + head] = logit;
        atomicMax(&menc[d * 2 + head], enc_f(logit));
    }
}

__global__ __launch_bounds__(256) void edge_accum(
    const float* __restrict__ vt, const int* __restrict__ esrc,
    const int* __restrict__ edst, const float* __restrict__ elog,
    const unsigned* __restrict__ menc, float* __restrict__ agg,
    float* __restrict__ sden, int E)
{
    int idx = blockIdx.x * blockDim.x + threadIdx.x;
    if (idx >= E * 128) return;
    int i = idx >> 7, c = idx & 127, h = c >> 6;
    int s = esrc[i], d = edst[i];
    float m = dec_f(menc[d * 2 + h]);
    float e = expf(elog[i * 2 + h] - m);
    atomicAdd(&agg[(size_t)d * 128 + c], e * vt[(size_t)s * 128 + c]);
    if ((c & 63) == 0) atomicAdd(&sden[d * 2 + h], e);
}

// normalize + GELU + bf16 (feeds the Wout GEMM A operand directly)
__global__ void seg_norm_gelu(const float* __restrict__ agg, const float* __restrict__ sden,
                              bf16_t* __restrict__ out, int Nn) {
    int idx = blockIdx.x * blockDim.x + threadIdx.x;
    if (idx >= Nn * 128) return;
    int n = idx >> 7, h = (idx >> 6) & 1;
    float s = sden[n * 2 + h];
    float v = (s > 0.0f) ? agg[idx] / s : 0.0f;
    out[idx] = to_bf16(gelu_f(v));
}

// ---------------------------------- tx destination: exactly 2 incoming edges
__global__ __launch_bounds__(256) void tx_agg(
    const float* __restrict__ Q0,      // [NTX,128] f32
    const float* __restrict__ kt1, const float* __restrict__ vt1,
    const float* __restrict__ kt3, const float* __restrict__ vt3,
    const int* __restrict__ esrc, const int* __restrict__ ecd,
    const int* __restrict__ edd,
    const float* __restrict__ prel, int p1off, int p3off,
    bf16_t* __restrict__ agg0b, int E)
{
    int i = blockIdx.x * 8 + (threadIdx.x >> 5);
    if (i >= E) return;
    int lane = threadIdx.x & 31, head = lane >> 4, t = lane & 15;
    int tx = esrc[i], c = ecd[i], d = edd[i];
    const float* q  = Q0 + (size_t)tx * 128 + head * 64;
    const float* k1 = kt1 + (size_t)c * 128 + head * 64;
    const float* k3 = kt3 + (size_t)d * 128 + head * 64;
    float p1 = 0.0f, p3 = 0.0f;
#pragma unroll
    for (int jj = 0; jj < 64; jj += 16) {
        float qv = q[jj + t];
        p1 += qv * k1[jj + t];
        p3 += qv * k3[jj + t];
    }
#pragma unroll
    for (int mk = 1; mk < 16; mk <<= 1) {
        p1 += __shfl_xor(p1, mk, 32);
        p3 += __shfl_xor(p3, mk, 32);
    }
    float l1 = p1 * prel[p1off + head] * 0.125f;
    float l3 = p3 * prel[p3off + head] * 0.125f;
    float mx = fmaxf(l1, l3);
    float e1 = expf(l1 - mx), e3 = expf(l3 - mx);
    float inv = 1.0f / (e1 + e3);
    float a1 = e1 * inv, a3 = e3 * inv;
    const float* v1 = vt1 + (size_t)c * 128 + head * 64;
    const float* v3 = vt3 + (size_t)d * 128 + head * 64;
    bf16_t* o = agg0b + (size_t)tx * 128 + head * 64;
#pragma unroll
    for (int jj = 0; jj < 64; jj += 16)
        o[jj + t] = to_bf16(gelu_f(a1 * v1[jj + t] + a3 * v3[jj + t]));
}

// ---------------------------------------------------------------- classifier
__global__ __launch_bounds__(256) void cls_kernel(
    const float* __restrict__ h0, const float* __restrict__ Wcls,
    const float* __restrict__ bcls, float* __restrict__ out, int M)
{
    int i = blockIdx.x * 8 + (threadIdx.x >> 5);
    if (i >= M) return;
    int lane = threadIdx.x & 31;
    float p = 0.0f;
    for (int j = lane; j < 128; j += 32) p += h0[(size_t)i * 128 + j] * Wcls[j];
#pragma unroll
    for (int mk = 16; mk >= 1; mk >>= 1) p += __shfl_xor(p, mk, 32);
    if (lane == 0) out[i] = p + bcls[0];
}

// =========================================================================
extern "C" void kernel_launch(void* const* d_in, const int* in_sizes, int n_in,
                              void* d_out, int out_size, void* d_ws, size_t ws_size,
                              hipStream_t stream) {
    const float* x_tx      = (const float*)d_in[0];
    const float* x_card    = (const float*)d_in[1];
    const float* x_dev     = (const float*)d_in[2];
    const float* decay     = (const float*)d_in[3];
    const int*   e_src     = (const int*)d_in[4];
    const int*   ec_dst    = (const int*)d_in[5];
    const int*   ed_dst    = (const int*)d_in[6];
    const float* Wlin_tx   = (const float*)d_in[7];
    const float* blin_tx   = (const float*)d_in[8];
    const float* Wlin_card = (const float*)d_in[9];
    const float* blin_card = (const float*)d_in[10];
    const float* Wlin_dev  = (const float*)d_in[11];
    const float* blin_dev  = (const float*)d_in[12];
    const float* Wkqv      = (const float*)d_in[13];
    const float* bkqv      = (const float*)d_in[14];
    const float* a_rel     = (const float*)d_in[15];
    const float* m_rel     = (const float*)d_in[16];
    const float* p_rel     = (const float*)d_in[17];
    const float* Wout      = (const float*)d_in[18];
    const float* bout      = (const float*)d_in[19];
    const float* skip      = (const float*)d_in[20];
    const float* Wcls      = (const float*)d_in[21];
    const float* bcls      = (const float*)d_in[22];

    // ---------------- workspace layout
    char* base = (char*)d_ws;
    size_t off = 0;
    auto alloc = [&](size_t bytes) -> void* {
        void* p = base + off;
        off += (bytes + 255) & ~(size_t)255;
        return p;
    };
    const int Ns[3] = {NTX, NCARD, NDEV};

    bf16_t* Wlin_p = (bf16_t*)alloc((size_t)128 * 128 * 2);
    bf16_t* Wkqv_p = (bf16_t*)alloc((size_t)6 * 128 * 384 * 2);
    bf16_t* Wout_p = (bf16_t*)alloc((size_t)6 * 128 * 128 * 2);
    bf16_t* Ablk_p = (bf16_t*)alloc((size_t)8 * 128 * 128 * 2);
    bf16_t* Mblk_p = (bf16_t*)alloc((size_t)8 * 128 * 128 * 2);
    bf16_t* x_txb  = (bf16_t*)alloc((size_t)NTX * 128 * 2);
    float *hA[3], *hB[3];
    bf16_t *hAb[3], *hBb[3];
    for (int i = 0; i < 3; ++i) hA[i]  = (float*)alloc((size_t)Ns[i] * 128 * 4);
    for (int i = 0; i < 3; ++i) hB[i]  = (float*)alloc((size_t)Ns[i] * 128 * 4);
    for (int i = 0; i < 3; ++i) hAb[i] = (bf16_t*)alloc((size_t)Ns[i] * 128 * 2);
    for (int i = 0; i < 3; ++i) hBb[i] = (bf16_t*)alloc((size_t)Ns[i] * 128 * 2);
    bf16_t* KQVb[3];   // [N,256]: K cols 0..127, V cols 128..255
    float*  Qf[3];     // [N,128] f32
    for (int i = 0; i < 3; ++i) KQVb[i] = (bf16_t*)alloc((size_t)Ns[i] * 256 * 2);
    for (int i = 0; i < 3; ++i) Qf[i]   = (float*)alloc((size_t)Ns[i] * 128 * 4);
    float* T1     = (float*)alloc((size_t)NTX * 128 * 4);
    float* T2     = (float*)alloc((size_t)NTX * 128 * 4);
    float* kt1buf = (float*)alloc((size_t)NCARD * 128 * 4);
    float* vt1buf = (float*)alloc((size_t)NCARD * 128 * 4);
    float* kt3buf = (float*)alloc((size_t)NDEV * 128 * 4);
    float* vt3buf = (float*)alloc((size_t)NDEV * 128 * 4);
    float* agg1   = (float*)alloc((size_t)NCARD * 128 * 4);
    float* agg2   = (float*)alloc((size_t)NDEV * 128 * 4);
    bf16_t* agg0b = (bf16_t*)alloc((size_t)NTX * 128 * 2);
    bf16_t* agg1b = (bf16_t*)alloc((size_t)NCARD * 128 * 2);
    bf16_t* agg2b = (bf16_t*)alloc((size_t)NDEV * 128 * 2);
    float* elog   = (float*)alloc((size_t)NTX * 2 * 4);
    unsigned* mencC = (unsigned*)alloc((size_t)NCARD * 2 * 4);
    float*    sdenC = (float*)alloc((size_t)NCARD * 2 * 4);
    unsigned* mencD = (unsigned*)alloc((size_t)NDEV * 2 * 4);
    float*    sdenD = (float*)alloc((size_t)NDEV * 2 * 4);

    auto cdiv = [](int a, int b) { return (a + b - 1) / b; };

    // ---------------- weight prep: bf16 fragment packing
    {
        int n = 128 * 128;
        pack_B<<<cdiv(n, 256), 256, 0, stream>>>(Wlin_tx, Wlin_p, 128);
        for (int mat = 0; mat < 6; ++mat) {
            int nk = 128 * 384;
            pack_B<<<cdiv(nk, 256), 256, 0, stream>>>(
                Wkqv + (size_t)mat * 128 * 384, Wkqv_p + (size_t)mat * 128 * 384, 384);
            pack_B<<<cdiv(n, 256), 256, 0, stream>>>(
                Wout + (size_t)mat * 128 * 128, Wout_p + (size_t)mat * 128 * 128, 128);
        }
        int nb = 8 * 128 * 128;
        pack_blockdiag_B<<<cdiv(nb, 256), 256, 0, stream>>>(a_rel, Ablk_p, 8);
        pack_blockdiag_B<<<cdiv(nb, 256), 256, 0, stream>>>(m_rel, Mblk_p, 8);
        cvt_bf16<<<cdiv(NTX * 128, 256), 256, 0, stream>>>(x_tx, x_txb, NTX * 128);
    }

    // ---------------- input projections
    wmma_gemm_bf16<1><<<cdiv(NTX, 64), 256, 0, stream>>>(
        x_txb, 128, Wlin_p, blin_tx, hA[0], hAb[0], nullptr, decay, nullptr, NTX, 128);
    lin_small<<<cdiv(NCARD * 128, 256), 256, 0, stream>>>(x_card, Wlin_card, blin_card,
                                                          hA[1], hAb[1], NCARD);
    lin_small<<<cdiv(NDEV * 128, 256), 256, 0, stream>>>(x_dev, Wlin_dev, blin_dev,
                                                         hA[2], hAb[2], NDEV);

    // ---------------- two HGT layers
    for (int l = 0; l < 2; ++l) {
        float**  IN   = (l == 0) ? hA : hB;
        float**  OUT  = (l == 0) ? hB : hA;
        bf16_t** INb  = (l == 0) ? hAb : hBb;
        bf16_t** OUTb = (l == 0) ? hBb : hAb;

        // KQV GEMMs: K/V -> bf16 KQVb, Q -> f32 Qf
        for (int i = 0; i < 3; ++i) {
            wmma_gemm_bf16<2><<<cdiv(Ns[i], 64), 256, 0, stream>>>(
                INb[i], 128, Wkqv_p + (size_t)(l * 3 + i) * 128 * 384,
                bkqv + (size_t)(l * 3 + i) * 384, Qf[i], KQVb[i],
                nullptr, nullptr, nullptr, Ns[i], 384);
        }

        // segment state init (ws is not re-poisoned between replays)
        fill_u32<<<cdiv(NCARD * 2, 256), 256, 0, stream>>>(mencC, ENC_NEGINF, NCARD * 2);
        fill_u32<<<cdiv(NCARD * 2, 256), 256, 0, stream>>>((unsigned*)sdenC, 0u, NCARD * 2);
        fill_u32<<<cdiv(NCARD * 128, 256), 256, 0, stream>>>((unsigned*)agg1, 0u, NCARD * 128);
        fill_u32<<<cdiv(NDEV * 2, 256), 256, 0, stream>>>(mencD, ENC_NEGINF, NDEV * 2);
        fill_u32<<<cdiv(NDEV * 2, 256), 256, 0, stream>>>((unsigned*)sdenD, 0u, NDEV * 2);
        fill_u32<<<cdiv(NDEV * 128, 256), 256, 0, stream>>>((unsigned*)agg2, 0u, NDEV * 128);
        fill_u32<<<cdiv(NTX * 64, 256), 256, 0, stream>>>((unsigned*)agg0b, 0u, NTX * 64);

        // relation 0: tx -> card
        wmma_gemm_bf16<0><<<cdiv(NTX, 64), 256, 0, stream>>>(
            KQVb[0] + 0, 256, Ablk_p + (size_t)(l * 4 + 0) * 16384, nullptr,
            T1, nullptr, nullptr, nullptr, nullptr, NTX, 128);
        wmma_gemm_bf16<0><<<cdiv(NTX, 64), 256, 0, stream>>>(
            KQVb[0] + 128, 256, Mblk_p + (size_t)(l * 4 + 0) * 16384, nullptr,
            T2, nullptr, nullptr, nullptr, nullptr, NTX, 128);
        edge_logits_max<<<cdiv(NTX, 8), 256, 0, stream>>>(
            Qf[1], T1, e_src, ec_dst, p_rel, (l * 4 + 0) * 2, elog, mencC, NTX);
        edge_accum<<<cdiv(NTX * 128, 256), 256, 0, stream>>>(
            T2, e_src, ec_dst, elog, mencC, agg1, sdenC, NTX);

        // relation 2: tx -> dev
        wmma_gemm_bf16<0><<<cdiv(NTX, 64), 256, 0, stream>>>(
            KQVb[0] + 0, 256, Ablk_p + (size_t)(l * 4 + 2) * 16384, nullptr,
            T1, nullptr, nullptr, nullptr, nullptr, NTX, 128);
        wmma_gemm_bf16<0><<<cdiv(NTX, 64), 256, 0, stream>>>(
            KQVb[0] + 128, 256, Mblk_p + (size_t)(l * 4 + 2) * 16384, nullptr,
            T2, nullptr, nullptr, nullptr, nullptr, NTX, 128);
        edge_logits_max<<<cdiv(NTX, 8), 256, 0, stream>>>(
            Qf[2], T1, e_src, ed_dst, p_rel, (l * 4 + 2) * 2, elog, mencD, NTX);
        edge_accum<<<cdiv(NTX * 128, 256), 256, 0, stream>>>(
            T2, e_src, ed_dst, elog, mencD, agg2, sdenD, NTX);

        // relations 1 (card -> tx) and 3 (dev -> tx)
        wmma_gemm_bf16<0><<<cdiv(NCARD, 64), 256, 0, stream>>>(
            KQVb[1] + 0, 256, Ablk_p + (size_t)(l * 4 + 1) * 16384, nullptr,
            kt1buf, nullptr, nullptr, nullptr, nullptr, NCARD, 128);
        wmma_gemm_bf16<0><<<cdiv(NCARD, 64), 256, 0, stream>>>(
            KQVb[1] + 128, 256, Mblk_p + (size_t)(l * 4 + 1) * 16384, nullptr,
            vt1buf, nullptr, nullptr, nullptr, nullptr, NCARD, 128);
        wmma_gemm_bf16<0><<<cdiv(NDEV, 64), 256, 0, stream>>>(
            KQVb[2] + 0, 256, Ablk_p + (size_t)(l * 4 + 3) * 16384, nullptr,
            kt3buf, nullptr, nullptr, nullptr, nullptr, NDEV, 128);
        wmma_gemm_bf16<0><<<cdiv(NDEV, 64), 256, 0, stream>>>(
            KQVb[2] + 128, 256, Mblk_p + (size_t)(l * 4 + 3) * 16384, nullptr,
            vt3buf, nullptr, nullptr, nullptr, nullptr, NDEV, 128);

        // tx destination: closed-form 2-edge softmax, GELU, bf16
        tx_agg<<<cdiv(NTX, 8), 256, 0, stream>>>(
            Qf[0], kt1buf, vt1buf, kt3buf, vt3buf, e_src, ec_dst, ed_dst,
            p_rel, (l * 4 + 1) * 2, (l * 4 + 3) * 2, agg0b, NTX);

        // card/dev: normalize + GELU + bf16
        seg_norm_gelu<<<cdiv(NCARD * 128, 256), 256, 0, stream>>>(agg1, sdenC, agg1b, NCARD);
        seg_norm_gelu<<<cdiv(NDEV * 128, 256), 256, 0, stream>>>(agg2, sdenD, agg2b, NDEV);

        // node update GEMMs
        bf16_t* aggb[3] = {agg0b, agg1b, agg2b};
        for (int t = 0; t < 3; ++t) {
            wmma_gemm_bf16<3><<<cdiv(Ns[t], 64), 256, 0, stream>>>(
                aggb[t], 128, Wout_p + (size_t)(l * 3 + t) * 16384,
                bout + (size_t)(l * 3 + t) * 128, OUT[t], OUTb[t],
                IN[t], nullptr, skip + (l * 3 + t), Ns[t], 128);
        }
    }

    // ---------------- classifier: layer-1 output lives in hA
    cls_kernel<<<cdiv(NTX, 8), 256, 0, stream>>>(hA[0], Wcls, bcls, (float*)d_out, NTX);
}
